// CapsuleRouting_12257836663238
// MI455X (gfx1250) — matrix operand order repaired and checked by reading, not compiled, not run on hardware
//
#include <hip/hip_runtime.h>

#define EPS   1e-5f
#define BATCH 16
#define BCAPS 32
#define CCAPS 32
#define PPDIM 16
#define HW    196

typedef __attribute__((ext_vector_type(16))) _Float16 v16h;
typedef __attribute__((ext_vector_type(8)))  float    v8f;

// One workgroup per (b, y, x) position. 256 threads = 8 waves (wave32).
// Phase 1: votes u[B][C][16] via v_wmma_f32_16x16x32_f16, block-diagonal A. -> LDS
// Phase 2: 3 dynamic-routing iterations entirely on-chip (LDS + shuffles).
__global__ __launch_bounds__(256)
void caps_route_wmma(const float* __restrict__ p,
                     const float* __restrict__ W,
                     float* __restrict__ out)
{
    extern __shared__ float smem[];
    float* u_s  = smem;                         // 32*32*16 = 16384 f (64 KB)
    float* sred = smem + 16384;                 // 8*32*16  =  4096 f (16 KB)
    float* sfin = smem + 16384 + 4096;          // 512 f
    float* v_s  = smem + 16384 + 4096 + 512;    // 512 f

    const int t    = threadIdx.x;
    const int wv   = t >> 5;        // wave id 0..7 -> B group
    const int lane = t & 31;
    const int pos  = blockIdx.x;    // 0..3135
    const int bb   = pos / HW;
    const int yx   = pos % HW;

    // ---------------- Phase 1: votes via WMMA ----------------
    {
        const int g   = wv;                 // handles B in [4g, 4g+4)
        const int M   = lane & 15;          // row = bl*4 + i
        const int bl  = M >> 2;
        const int ii  = M & 3;
        const int B   = g * 4 + bl;
        const bool hi = (lane >= 16);

        // p4[b,B,i,j,y,x] = p[b,B,i*4+j,y,x]
        const float* prow = p + ((size_t)(bb * BCAPS + B) * PPDIM) * HW + yx;
        float pv[4];
        #pragma unroll
        for (int j = 0; j < 4; ++j) pv[j] = prow[(ii * 4 + j) * HW];

        // A (16x32 f16): lanes<16 carry K 0..7 in elems 0..7, lanes>=16 carry K 8..15.
        // Nonzero iff K>>2 == bl (block-diagonal over the 4 B-locals).
        v16h A = {};
        #pragma unroll
        for (int e = 0; e < 8; ++e) {
            const int K = e + (hi ? 8 : 0);        // K&3 == e&3 (compile-time)
            A[e] = ((K >> 2) == bl) ? (_Float16)pv[e & 3] : (_Float16)0.0f;
        }

        const int N = lane & 15;                   // B-matrix column
        for (int chunk = 0; chunk < 8; ++chunk) {
            // W flat: B*512 + j*128 + (k*32 + C); column ncol = chunk*16 + N
            const float* Wg = W + (size_t)g * 2048 + chunk * 16 + N;
            v16h Bf = {};
            #pragma unroll
            for (int e = 0; e < 8; ++e) {          // row K=e+8h: B'=4g+(K>>2), j=K&3
                const int K = e + (hi ? 8 : 0);
                const float wval = Wg[(K >> 2) * 512 + (K & 3) * 128];
                Bf[e] = (_Float16)wval;            // elems 8..15: K 16..31 = 0
            }
            v8f D = {};
            D = __builtin_amdgcn_wmma_f32_16x16x32_f16(
                    false, A, false, Bf, (short)0, D, false, false);

            const int ncol = chunk * 16 + N;
            const int kk   = ncol >> 5;
            const int C    = ncol & 31;
            #pragma unroll
            for (int r = 0; r < 8; ++r) {          // D: M = r + (hi?8:0), N = lane&15
                const int Mr = r + (hi ? 8 : 0);
                const int Br = g * 4 + (Mr >> 2);
                const int ik = (Mr & 3) * 4 + kk;
                u_s[((Br * 32 + C) << 4) + ik] = D[r];
            }
        }
    }
    __syncthreads();

    // ---------------- Phase 2: dynamic routing ----------------
    const int B  = t >> 3;          // this thread's input capsule
    const int C0 = (t & 7) * 4;     // its 4 output capsules

    float rr[4] = {0.f, 0.f, 0.f, 0.f};

    for (int iter = 0; iter < 3; ++iter) {
        // softmax over C: 32 values spread over 8 lanes (xor 1,2,4) x 4 regs
        float mx = fmaxf(fmaxf(rr[0], rr[1]), fmaxf(rr[2], rr[3]));
        mx = fmaxf(mx, __shfl_xor(mx, 1, 32));
        mx = fmaxf(mx, __shfl_xor(mx, 2, 32));
        mx = fmaxf(mx, __shfl_xor(mx, 4, 32));
        float ex[4], es = 0.f;
        #pragma unroll
        for (int c2 = 0; c2 < 4; ++c2) { ex[c2] = __expf(rr[c2] - mx); es += ex[c2]; }
        es += __shfl_xor(es, 1, 32);
        es += __shfl_xor(es, 2, 32);
        es += __shfl_xor(es, 4, 32);
        const float inv = 1.f / es;

        // partial s = c * u ; reduce over the wave's 4 B (xor 8,16)
        float sp[64];
        #pragma unroll
        for (int c2 = 0; c2 < 4; ++c2) {
            const float cc = ex[c2] * inv;
            const float* ur = &u_s[((B * 32 + C0 + c2) << 4)];
            #pragma unroll
            for (int q = 0; q < 16; ++q) sp[c2 * 16 + q] = cc * ur[q];
        }
        #pragma unroll
        for (int k2 = 0; k2 < 64; ++k2) {
            sp[k2] += __shfl_xor(sp[k2], 8, 32);
            sp[k2] += __shfl_xor(sp[k2], 16, 32);
        }
        if ((lane >> 3) == 0) {                    // lanes 0..7: one writer per C-group
            #pragma unroll
            for (int c2 = 0; c2 < 4; ++c2)
                #pragma unroll
                for (int q = 0; q < 16; ++q)
                    sred[((wv * 32 + C0 + c2) << 4) + q] = sp[c2 * 16 + q];
        }
        __syncthreads();

        // cross-wave reduce: 512 s-elements, 2 per thread
        #pragma unroll
        for (int u2 = 0; u2 < 2; ++u2) {
            const int e2 = 2 * t + u2;
            float acc = 0.f;
            #pragma unroll
            for (int w8 = 0; w8 < 8; ++w8) acc += sred[(w8 << 9) + e2];
            sfin[e2] = acc;
        }
        __syncthreads();

        // squash: v = s * norm/(1+norm^2), norm^2 = sum(s^2)+EPS  (one thread per C)
        if (t < 32) {
            float nsq = EPS;
            #pragma unroll
            for (int q = 0; q < 16; ++q) { const float s = sfin[(t << 4) + q]; nsq += s * s; }
            const float nrm = sqrtf(nsq);
            const float f   = nrm / (1.f + nsq);
            #pragma unroll
            for (int q = 0; q < 16; ++q) v_s[(t << 4) + q] = sfin[(t << 4) + q] * f;
        }
        __syncthreads();

        if (iter < 2) {                            // agreement update r += <u, v>
            #pragma unroll
            for (int c2 = 0; c2 < 4; ++c2) {
                const float* ur = &u_s[((B * 32 + C0 + c2) << 4)];
                const float* vr = &v_s[((C0 + c2) << 4)];
                float dot = 0.f;
                #pragma unroll
                for (int q = 0; q < 16; ++q) dot += ur[q] * vr[q];
                rr[c2] += dot;
            }
        }
    }

    // ---------------- outputs ----------------
    // v: (b, C, 16, h, w)
    #pragma unroll
    for (int u2 = 0; u2 < 2; ++u2) {
        const int e2 = 2 * t + u2;
        const int C  = e2 >> 4;
        const int ik = e2 & 15;
        out[((size_t)(bb * CCAPS + C) * PPDIM + ik) * HW + yx] = v_s[e2];
    }
    // a_out: (b, C, h, w) = sqrt(sum(v^2)+EPS), appended after v
    if (t < 32) {
        float asq = EPS;
        #pragma unroll
        for (int q = 0; q < 16; ++q) { const float vq = v_s[(t << 4) + q]; asq += vq * vq; }
        out[(size_t)BATCH * CCAPS * PPDIM * HW + (size_t)(bb * CCAPS + t) * HW + yx] = sqrtf(asq);
    }
}

extern "C" void kernel_launch(void* const* d_in, const int* in_sizes, int n_in,
                              void* d_out, int out_size, void* d_ws, size_t ws_size,
                              hipStream_t stream) {
    const float* p = (const float*)d_in[0];
    // d_in[1] ("a") is unused by the reference computation
    const float* W = (const float*)d_in[2];
    float* out = (float*)d_out;

    const dim3 grid(BATCH * HW);   // 3136 positions
    const dim3 block(256);         // 8 waves
    const size_t shmem = (size_t)(16384 + 4096 + 512 + 512) * sizeof(float); // 84 KB < 320 KB/WGP
    hipLaunchKernelGGL(caps_route_wmma, grid, block, shmem, stream, p, W, out);
}
